// MetaBaseline_80865644249958
// MI455X (gfx1250) — matrix operand order repaired and checked by reading, not compiled, not run on hardware
//
#include <hip/hip_runtime.h>
#include <hip/hip_bf16.h>

typedef __attribute__((ext_vector_type(16))) __bf16 v16bf;
typedef __attribute__((ext_vector_type(8)))  float  v8f;

constexpr int B_   = 64;
constexpr int WAY  = 5;
constexpr int D_   = 640;
constexpr int HW   = 25;
constexpr int NQ   = 75;
constexpr int M_   = NQ * HW;        // 1875 query local descriptors per image
constexpr int MT   = (M_ + 15) / 16; // 118 M-tiles
constexpr int NPAD = 128;            // 125 support cols padded to 8 N-tiles

// ---------------- K1: support prep ----------------
// For each (b, way): column-normalize [D, hw] over D, store TRANSPOSED bf16
// as sn[b][n][k] (k contiguous -> 32B-aligned WMMA B-fragment loads), and
// write the L2-normalized spatial-mean descriptor bm[b][way][D].
__global__ void k_support_prep(const float* __restrict__ base,
                               __bf16* __restrict__ sn,
                               float* __restrict__ bm)
{
    int b = blockIdx.y, k = blockIdx.x, t = threadIdx.x;
    const float* x = base + (size_t)(b * WAY + k) * D_ * HW;

    __shared__ float redbuf[256];
    __shared__ float colinv[HW];
    __shared__ float meanv[D_];
    __shared__ float minv;

    float cs[HW];
    #pragma unroll
    for (int l = 0; l < HW; ++l) cs[l] = 0.f;
    for (int d = t; d < D_; d += 256) {
        #pragma unroll
        for (int l = 0; l < HW; ++l) { float v = x[d * HW + l]; cs[l] += v * v; }
    }
    for (int l = 0; l < HW; ++l) {
        redbuf[t] = cs[l]; __syncthreads();
        for (int s = 128; s > 0; s >>= 1) { if (t < s) redbuf[t] += redbuf[t + s]; __syncthreads(); }
        if (t == 0) colinv[l] = rsqrtf(fmaxf(redbuf[0], 1e-24f));
        __syncthreads();
    }

    float msq = 0.f;
    for (int d = t; d < D_; d += 256) {
        float s = 0.f;
        #pragma unroll
        for (int l = 0; l < HW; ++l) s += x[d * HW + l];
        s *= (1.f / HW);
        meanv[d] = s; msq += s * s;
    }
    redbuf[t] = msq; __syncthreads();
    for (int s = 128; s > 0; s >>= 1) { if (t < s) redbuf[t] += redbuf[t + s]; __syncthreads(); }
    if (t == 0) minv = 1.f / fmaxf(sqrtf(redbuf[0]), 1e-12f);
    __syncthreads();
    for (int d = t; d < D_; d += 256) bm[(size_t)(b * WAY + k) * D_ + d] = meanv[d] * minv;

    __bf16* snb = sn + (size_t)b * NPAD * D_;
    for (int e = t; e < D_ * HW; e += 256) {
        int d = e / HW, l = e % HW;
        snb[(size_t)(k * HW + l) * D_ + d] = (__bf16)(x[d * HW + l] * colinv[l]);
    }
    if (k == 0) {  // zero pad rows n = 125..127 once per b
        for (int e = t; e < (NPAD - WAY * HW) * D_; e += 256)
            snb[(size_t)(WAY * HW) * D_ + e] = (__bf16)0.f;
    }
}

// ---------------- K2: query prep + global-cosine init ----------------
// Per (b, q): inverse patch norms qinv, normalized query mean, and
// d_out[b,q,:] = r0 * (qm . bm)  (initializes the output).
__global__ void k_query_prep(const float* __restrict__ query,
                             const float* __restrict__ bm,
                             const float* __restrict__ r,
                             float* __restrict__ qinv,
                             float* __restrict__ out)
{
    int b = blockIdx.y, q = blockIdx.x, t = threadIdx.x;
    const float* x = query + (size_t)(b * NQ + q) * D_ * HW;

    __shared__ float redbuf[256];
    __shared__ float meanv[D_];
    __shared__ float minv;

    float cs[HW];
    #pragma unroll
    for (int l = 0; l < HW; ++l) cs[l] = 0.f;
    for (int d = t; d < D_; d += 256) {
        #pragma unroll
        for (int l = 0; l < HW; ++l) { float v = x[d * HW + l]; cs[l] += v * v; }
    }
    for (int l = 0; l < HW; ++l) {
        redbuf[t] = cs[l]; __syncthreads();
        for (int s = 128; s > 0; s >>= 1) { if (t < s) redbuf[t] += redbuf[t + s]; __syncthreads(); }
        if (t == 0) qinv[(size_t)b * M_ + q * HW + l] = rsqrtf(fmaxf(redbuf[0], 1e-24f));
        __syncthreads();
    }

    float msq = 0.f;
    for (int d = t; d < D_; d += 256) {
        float s = 0.f;
        #pragma unroll
        for (int l = 0; l < HW; ++l) s += x[d * HW + l];
        s *= (1.f / HW);
        meanv[d] = s; msq += s * s;
    }
    redbuf[t] = msq; __syncthreads();
    for (int s = 128; s > 0; s >>= 1) { if (t < s) redbuf[t] += redbuf[t + s]; __syncthreads(); }
    if (t == 0) minv = 1.f / fmaxf(sqrtf(redbuf[0]), 1e-12f);  // torch F.normalize eps
    __syncthreads();

    float dk[WAY];
    #pragma unroll
    for (int k = 0; k < WAY; ++k) dk[k] = 0.f;
    for (int d = t; d < D_; d += 256) {
        float mv = meanv[d] * minv;
        #pragma unroll
        for (int k = 0; k < WAY; ++k) dk[k] += mv * bm[(size_t)(b * WAY + k) * D_ + d];
    }
    float r0 = r[0];
    for (int k = 0; k < WAY; ++k) {
        redbuf[t] = dk[k]; __syncthreads();
        for (int s = 128; s > 0; s >>= 1) { if (t < s) redbuf[t] += redbuf[t + s]; __syncthreads(); }
        if (t == 0) out[(size_t)(b * NQ + q) * WAY + k] = r0 * redbuf[0];
        __syncthreads();
    }
}

// ---------------- K3: bf16 WMMA GEMM + fused top-k epilogue ----------------
// Wave = one 16-row M-tile x all 128 padded support columns, K = 640 in
// 20 steps of 32 using v_wmma_f32_16x16x32_bf16 (fp32 accumulate).
__global__ void __launch_bounds__(256)
k_gemm(const float* __restrict__ query,
       const __bf16* __restrict__ sn,
       const float* __restrict__ qinv,
       const float* __restrict__ r,
       const int* __restrict__ nkp,
       float* __restrict__ out)
{
    __shared__ float tile[8][16][NPAD];  // 64 KB (<< 320 KB/WGP)

    int b    = blockIdx.y;
    int wave = threadIdx.x >> 5;
    int lane = threadIdx.x & 31;
    int half = lane >> 4;
    int r16  = lane & 15;
    int mtile = blockIdx.x * 8 + wave;
    bool wactive = (mtile < MT);

    int Mrow = mtile * 16 + r16;               // A-matrix row owned by this lane
    bool vrow = wactive && (Mrow < M_);
    int Ms = vrow ? Mrow : 0;
    int qi = Ms / HW, p = Ms % HW;
    const float* qb = query + (size_t)(b * NQ + qi) * D_ * HW + p;  // stride HW per k
    float invn = vrow ? qinv[(size_t)b * M_ + Mrow] : 0.f;

    const __bf16* snb = sn + (size_t)b * NPAD * D_;

    v8f acc[8];
    #pragma unroll
    for (int nt = 0; nt < 8; ++nt)
        #pragma unroll
        for (int j = 0; j < 8; ++j) acc[nt][j] = 0.f;

    if (wactive) {                              // wave-uniform: EXEC all-ones for WMMA
        for (int k0 = 0; k0 < D_; k0 += 32) {
            int ka = k0 + half * 16;
            // A fragment, contiguous K order (matches B):
            //   elems 0..7  -> K = k0 + 16*half + i
            //   elems 8..15 -> K = k0 + 16*half + 8 + i
            union { v16bf v; __bf16 e[16]; } A;
            #pragma unroll
            for (int i = 0; i < 8; ++i) {
                A.e[i]     = (__bf16)(qb[(size_t)(ka + i) * HW]      * invn);
                A.e[8 + i] = (__bf16)(qb[(size_t)(ka + 8 + i) * HW] * invn);
            }
            #pragma unroll
            for (int nt = 0; nt < 8; ++nt) {
                // B fragment: column n = nt*16 + r16, 16 contiguous K values
                // (lanes 0-15: K=k0..k0+15, lanes 16-31: K=k0+16..k0+31) = one 32B load
                const __bf16* bp = snb + (size_t)(nt * 16 + r16) * D_ + k0 + half * 16;
                v16bf Bv = *(const v16bf*)bp;
                acc[nt] = __builtin_amdgcn_wmma_f32_16x16x32_bf16(
                    false, A.v, false, Bv, (short)0, acc[nt], false, false);
            }
        }
        // D layout: row = j + 8*half, col-in-tile = lane&15
        #pragma unroll
        for (int nt = 0; nt < 8; ++nt)
            #pragma unroll
            for (int j = 0; j < 8; ++j)
                tile[wave][j + 8 * half][nt * 16 + r16] = acc[nt][j];
    }
    __syncthreads();

    if (wactive) {
        int nk = *nkp; if (nk > HW) nk = HW; if (nk < 1) nk = 1;
        float scale = r[1] / (float)nk;
        for (int seg = lane; seg < 16 * WAY; seg += 32) {
            int row = seg / WAY, w = seg % WAY;
            int Mg = mtile * 16 + row;
            if (Mg >= M_) continue;
            float vals[HW];
            #pragma unroll
            for (int l = 0; l < HW; ++l) vals[l] = tile[wave][row][w * HW + l];
            float s = 0.f;
            for (int t2 = 0; t2 < nk; ++t2) {   // top-k via repeated max, reg-resident
                int bi = 0; float bv = vals[0];
                #pragma unroll
                for (int l = 1; l < HW; ++l) if (vals[l] > bv) { bv = vals[l]; bi = l; }
                s += bv;
                #pragma unroll
                for (int l = 0; l < HW; ++l) vals[l] = (l == bi) ? -3.0e38f : vals[l];
            }
            int qg = Mg / HW;
            atomicAdd(&out[(size_t)(b * NQ + qg) * WAY + w], s * scale);
        }
    }
}

extern "C" void kernel_launch(void* const* d_in, const int* in_sizes, int n_in,
                              void* d_out, int out_size, void* d_ws, size_t ws_size,
                              hipStream_t stream)
{
    const float* base  = (const float*)d_in[0];
    const float* query = (const float*)d_in[1];
    const float* r     = (const float*)d_in[2];
    const int*   nk    = (const int*)d_in[3];
    float* out = (float*)d_out;

    char* ws = (char*)d_ws;
    __bf16* sn  = (__bf16*)ws;                                   // 64*128*640*2 = 10485760 B
    float*  bm  = (float*)(ws + 10485760);                       // 64*5*640*4   =   819200 B
    float*  qinv= (float*)(ws + 10485760 + 819200);              // 64*1875*4    =   480000 B

    k_support_prep<<<dim3(WAY, B_), 256, 0, stream>>>(base, sn, bm);
    k_query_prep <<<dim3(NQ,  B_), 256, 0, stream>>>(query, bm, r, qinv, out);
    k_gemm       <<<dim3((MT + 7) / 8, B_), 256, 0, stream>>>(query, sn, qinv, r, nk, out);
}
